// HybridLoss_11132555231787
// MI455X (gfx1250) — compile-verified, matrix operand order checked by reading, not run on hardware
//
#include <hip/hip_runtime.h>
#include <hip/hip_bf16.h>

typedef __attribute__((ext_vector_type(16))) _Float16 v16h;
typedef __attribute__((ext_vector_type(8)))  _Float16 v8h;
typedef __attribute__((ext_vector_type(8)))  float    v8f;

#define B_ROWS 8192
#define DIMD   1024
#define EPSN   1e-8f

// ---------------------------------------------------------------------------
// Workspace layout (bytes):
//   [0,                16M)   qn  f16 [8192][1024]  (normalized student query)
//   [16M,              32M)   pn  f16 [8192][1024]  (normalized student passage)
//   [32M,         32M+32K)    diag        f32 [8192]
//   [32M+32K,     32M+64K)    row_expsum  f32 [8192]
//   [32M+64K,  32M+64K+8)     mse_acc     f32 [2]
// ---------------------------------------------------------------------------

#if __has_builtin(__builtin_amdgcn_global_load_async_to_lds_b128)
#define HAVE_ASYNC_COPY 1
#else
#define HAVE_ASYNC_COPY 0
#endif

#if HAVE_ASYNC_COPY
// Builtin prototype (per hipcc diagnostic): first param is a pointer to a
// 128-bit int vector (vector_size(16)), global src first, LDS dst second,
// then two immediates (offset, cpol).
typedef int v4i_vs __attribute__((vector_size(16)));
typedef __attribute__((address_space(1))) v4i_vs* gptr_v4i;
typedef __attribute__((address_space(3))) v4i_vs* lptr_v4i;

__device__ __forceinline__ void async_cp_b128(_Float16* lds_dst, const _Float16* gsrc) {
  __builtin_amdgcn_global_load_async_to_lds_b128(
      (gptr_v4i)(void*)gsrc, (lptr_v4i)(void*)lds_dst, 0, 0);
}

template <int N>
__device__ __forceinline__ void wait_async() {
#if __has_builtin(__builtin_amdgcn_s_wait_asynccnt)
  __builtin_amdgcn_s_wait_asynccnt((unsigned short)N);
#else
  asm volatile("s_wait_asynccnt %0" ::"n"(N) : "memory");
#endif
}
#endif

__device__ __forceinline__ float blockReduceSum256(float v) {
  #pragma unroll
  for (int off = 16; off > 0; off >>= 1) v += __shfl_xor(v, off, 32);
  __shared__ float smem[8];
  const int lane = threadIdx.x & 31;
  const int wv   = threadIdx.x >> 5;
  __syncthreads();                 // protect smem reuse across calls
  if (lane == 0) smem[wv] = v;
  __syncthreads();
  float r = smem[0];
  #pragma unroll
  for (int w = 1; w < 8; w++) r += smem[w];
  return r;
}

__global__ __launch_bounds__(256)
void init_kernel(float* __restrict__ row_expsum, float* __restrict__ mse_acc) {
  const int idx = blockIdx.x * 256 + threadIdx.x;
  if (idx < B_ROWS) row_expsum[idx] = 0.0f;
  if (idx < 2)      mse_acc[idx]    = 0.0f;
}

// One block per row: norms, MSE partials, f16 normalized writes, diagonal dot.
__global__ __launch_bounds__(256)
void prep_kernel(const float* __restrict__ sq, const float* __restrict__ sp,
                 const float* __restrict__ tq, const float* __restrict__ tp,
                 _Float16* __restrict__ qn, _Float16* __restrict__ pn,
                 float* __restrict__ diag, float* __restrict__ mse_acc) {
  const int row = blockIdx.x;
  const size_t base = (size_t)row * DIMD;
  float qv[4], pv[4];
  float lq = 0.f, lp = 0.f, ldq = 0.f, ldp = 0.f;
  #pragma unroll
  for (int k = 0; k < 4; k++) {
    const int j = threadIdx.x + k * 256;
    const float a  = sq[base + j], b  = sp[base + j];
    const float ta = tq[base + j], tb = tp[base + j];
    qv[k] = a; pv[k] = b;
    lq += a * a; lp += b * b;
    const float d1 = a - ta; ldq += d1 * d1;
    const float d2 = b - tb; ldp += d2 * d2;
  }
  const float nq  = blockReduceSum256(lq);
  const float np  = blockReduceSum256(lp);
  const float sdq = blockReduceSum256(ldq);
  const float sdp = blockReduceSum256(ldp);
  if (threadIdx.x == 0) {
    atomicAdd(&mse_acc[0], sdq);
    atomicAdd(&mse_acc[1], sdp);
  }
  const float iq = 1.0f / fmaxf(sqrtf(nq), EPSN);
  const float ip = 1.0f / fmaxf(sqrtf(np), EPSN);
  float dd = 0.f;
  #pragma unroll
  for (int k = 0; k < 4; k++) {
    const int j = threadIdx.x + k * 256;
    const float a = qv[k] * iq, b = pv[k] * ip;
    qn[base + j] = (_Float16)a;
    pn[base + j] = (_Float16)b;
    dd += a * b;
  }
  const float dsum = blockReduceSum256(dd);
  if (threadIdx.x == 0) diag[row] = dsum;
}

// 128x128 logits tile per workgroup; 8 waves (4x2), 2x4 WMMA tiles per wave.
// Double-buffered LDS; async global->LDS copies (ASYNCcnt) overlap the next
// K-slice's loads with the current slice's WMMAs. Logits are cosine sims in
// [-1,1], so logsumexp needs no max pass: accumulate sum(exp) with atomics.
__global__ __launch_bounds__(256)
void cosine_gemm_kernel(const _Float16* __restrict__ qn,
                        const _Float16* __restrict__ pn,
                        float* __restrict__ row_expsum) {
  constexpr int KT  = 32;
  constexpr int LDA = 40;     // padded f16 stride: conflict-free b128 reads
  constexpr int NS  = DIMD / KT;
  __shared__ __align__(16) _Float16 As[2][128 * LDA];
  __shared__ __align__(16) _Float16 Bs[2][128 * LDA];

  const int tid  = threadIdx.x;
  const int wave = tid >> 5;
  const int lane = tid & 31;
  const int wm = wave & 3;                // 32-row strip within tile
  const int wn = wave >> 2;               // 64-col strip within tile
  const int rowBase = blockIdx.y * 128;
  const int colBase = blockIdx.x * 128;

  // staging: each thread moves 16 halves of A and 16 of B per K-slice
  const int srow = tid >> 1;
  const int scol = (tid & 1) * 16;
  const _Float16* gA = qn + (size_t)(rowBase + srow) * DIMD + scol;
  const _Float16* gB = pn + (size_t)(colBase + srow) * DIMD + scol;
  const int lidx = srow * LDA + scol;

  // fragment lane mapping (16-bit A/B layout, wave32)
  const int fr   = lane & 15;
  const int koff = (lane >> 4) * 8;

  v8f acc[2][4] = {};

#if HAVE_ASYNC_COPY
  // ---- software-pipelined path: async copies into double-buffered LDS ----
  {
    async_cp_b128(&As[0][lidx],     gA);
    async_cp_b128(&As[0][lidx + 8], gA + 8);
    async_cp_b128(&Bs[0][lidx],     gB);
    async_cp_b128(&Bs[0][lidx + 8], gB + 8);
  }
  for (int s = 0; s < NS; ++s) {
    const int cur = s & 1;
    if (s + 1 < NS) {
      const int k1 = (s + 1) * KT;
      async_cp_b128(&As[cur ^ 1][lidx],     gA + k1);
      async_cp_b128(&As[cur ^ 1][lidx + 8], gA + k1 + 8);
      async_cp_b128(&Bs[cur ^ 1][lidx],     gB + k1);
      async_cp_b128(&Bs[cur ^ 1][lidx + 8], gB + k1 + 8);
      wait_async<4>();   // slice s complete (in-order), slice s+1 in flight
    } else {
      wait_async<0>();
    }
    __syncthreads();

    v16h a[2], b[4];
    #pragma unroll
    for (int m = 0; m < 2; m++) {
      const _Float16* src = &As[cur][(wm * 32 + m * 16 + fr) * LDA + koff];
      const v8h lo = *(const v8h*)(src);
      const v8h hi = *(const v8h*)(src + 16);
      #pragma unroll
      for (int e = 0; e < 8; e++) { a[m][e] = lo[e]; a[m][8 + e] = hi[e]; }
    }
    #pragma unroll
    for (int n = 0; n < 4; n++) {
      const _Float16* src = &Bs[cur][(wn * 64 + n * 16 + fr) * LDA + koff];
      const v8h lo = *(const v8h*)(src);
      const v8h hi = *(const v8h*)(src + 16);
      #pragma unroll
      for (int e = 0; e < 8; e++) { b[n][e] = lo[e]; b[n][8 + e] = hi[e]; }
    }
    #pragma unroll
    for (int m = 0; m < 2; m++)
      #pragma unroll
      for (int n = 0; n < 4; n++)
        acc[m][n] = __builtin_amdgcn_wmma_f32_16x16x32_f16(
            false, a[m], false, b[n], (short)0, acc[m][n], false, false);
    __syncthreads();   // all reads of buf `cur` done before it is refilled
  }
#else
  // ---- fallback: synchronous staging, single logical pipeline ----
  for (int s = 0; s < NS; ++s) {
    const int k0 = s * KT;
    *(v8h*)(&As[0][lidx])     = *(const v8h*)(gA + k0);
    *(v8h*)(&As[0][lidx + 8]) = *(const v8h*)(gA + k0 + 8);
    *(v8h*)(&Bs[0][lidx])     = *(const v8h*)(gB + k0);
    *(v8h*)(&Bs[0][lidx + 8]) = *(const v8h*)(gB + k0 + 8);
    __syncthreads();

    v16h a[2], b[4];
    #pragma unroll
    for (int m = 0; m < 2; m++) {
      const _Float16* src = &As[0][(wm * 32 + m * 16 + fr) * LDA + koff];
      const v8h lo = *(const v8h*)(src);
      const v8h hi = *(const v8h*)(src + 16);
      #pragma unroll
      for (int e = 0; e < 8; e++) { a[m][e] = lo[e]; a[m][8 + e] = hi[e]; }
    }
    #pragma unroll
    for (int n = 0; n < 4; n++) {
      const _Float16* src = &Bs[0][(wn * 64 + n * 16 + fr) * LDA + koff];
      const v8h lo = *(const v8h*)(src);
      const v8h hi = *(const v8h*)(src + 16);
      #pragma unroll
      for (int e = 0; e < 8; e++) { b[n][e] = lo[e]; b[n][8 + e] = hi[e]; }
    }
    #pragma unroll
    for (int m = 0; m < 2; m++)
      #pragma unroll
      for (int n = 0; n < 4; n++)
        acc[m][n] = __builtin_amdgcn_wmma_f32_16x16x32_f16(
            false, a[m], false, b[n], (short)0, acc[m][n], false, false);
    __syncthreads();
  }
#endif

  // epilogue: sum exp over this wave's 64 columns, reduce across 16 lanes,
  // atomically accumulate per-row partial expsums.
  #pragma unroll
  for (int m = 0; m < 2; m++) {
    #pragma unroll
    for (int g = 0; g < 8; g++) {
      float s = 0.f;
      #pragma unroll
      for (int n = 0; n < 4; n++) s += __expf(acc[m][n][g]);
      #pragma unroll
      for (int off = 1; off < 16; off <<= 1) s += __shfl_xor(s, off, 32);
      if (fr == 0) {
        const int row = rowBase + wm * 32 + m * 16 + g + (lane >> 4) * 8;
        atomicAdd(&row_expsum[row], s);
      }
    }
  }
}

__global__ __launch_bounds__(256)
void finalize_kernel(const float* __restrict__ row_expsum,
                     const float* __restrict__ diag,
                     const float* __restrict__ mse_acc,
                     float* __restrict__ out) {
  float local = 0.f;
  for (int j = threadIdx.x; j < B_ROWS; j += 256)
    local += logf(row_expsum[j]) - diag[j];
  const float total = blockReduceSum256(local);
  if (threadIdx.x == 0) {
    const float retrieval = total / (float)B_ROWS;
    const float distill =
        0.5f * (mse_acc[0] + mse_acc[1]) / (float)((size_t)B_ROWS * DIMD);
    out[0] = 0.5f * distill + 0.5f * retrieval;
  }
}

extern "C" void kernel_launch(void* const* d_in, const int* in_sizes, int n_in,
                              void* d_out, int out_size, void* d_ws, size_t ws_size,
                              hipStream_t stream) {
  const float* sq = (const float*)d_in[0];   // student_query_emb
  const float* sp = (const float*)d_in[1];   // student_passage_emb
  const float* tq = (const float*)d_in[2];   // teacher_query_emb
  const float* tp = (const float*)d_in[3];   // teacher_passage_emb

  char* ws = (char*)d_ws;
  _Float16* qn = (_Float16*)ws;
  _Float16* pn = (_Float16*)(ws + (size_t)B_ROWS * DIMD * 2);
  float* diag       = (float*)(ws + (size_t)B_ROWS * DIMD * 4);
  float* row_expsum = diag + B_ROWS;
  float* mse_acc    = row_expsum + B_ROWS;

  init_kernel<<<(B_ROWS + 255) / 256, 256, 0, stream>>>(row_expsum, mse_acc);
  prep_kernel<<<B_ROWS, 256, 0, stream>>>(sq, sp, tq, tp, qn, pn, diag, mse_acc);
  dim3 grid(B_ROWS / 128, B_ROWS / 128);
  cosine_gemm_kernel<<<grid, 256, 0, stream>>>(qn, pn, row_expsum);
  finalize_kernel<<<1, 256, 0, stream>>>(row_expsum, diag, mse_acc, (float*)d_out);
}
